// VQ_VAE_67190468378651
// MI455X (gfx1250) — matrix operand order repaired and checked by reading, not compiled, not run on hardware
//
#include <hip/hip_runtime.h>

typedef __attribute__((ext_vector_type(16))) _Float16 v16h;
typedef __attribute__((ext_vector_type(8)))  float    v8f;

#define BTOT 8192
#define LDIM 512
#define DDIM 64
#define KCB  512
#define YDIM 100

// ===========================================================================
// A-fragment packing (once per launch). Contraction order: k = tap*32 + ic.
// WMMA 16-bit A 16x32 layout: lanes 0-15/16-31 both carry M=lane&15;
// this code packs half j -> K = j + (hi?16:0).  Fragment f: [f][lane][16 halfs].
// ===========================================================================
__global__ void pack_a_enc_kernel(const float* __restrict__ w,  // [OC][IC][4]
                                  _Float16* __restrict__ af,
                                  int OC, int IC, int K, int ksteps, int mtiles) {
  const int f = blockIdx.x * (blockDim.x >> 5) + (threadIdx.x >> 5);
  if (f >= mtiles * ksteps) return;
  const int mt = f / ksteps, ks = f % ksteps;
  const int lane = threadIdx.x & 31;
  const bool hi = lane >= 16;
  const int m = min(mt * 16 + (lane & 15), OC - 1);   // clamp: dup rows, never stored
#pragma unroll
  for (int j = 0; j < 16; ++j) {
    int ka = ks * 32 + j + (hi ? 16 : 0);
    int t = ka >> 5, ic = ka & 31;
    float v = (ka < K) ? w[(m * IC + ic) * 4 + t] : 0.0f;
    af[((size_t)f * 32 + lane) * 16 + j] = (_Float16)v;
  }
}

__global__ void pack_a_dec_kernel(const float* __restrict__ w,  // [IC][OC][4]
                                  _Float16* __restrict__ af,
                                  int OC, int K, int ksteps, int mtiles) {
  const int f = blockIdx.x * (blockDim.x >> 5) + (threadIdx.x >> 5);
  if (f >= 2 * mtiles * ksteps) return;
  const int par = f / (mtiles * ksteps);
  const int r = f % (mtiles * ksteps);
  const int mt = r / ksteps, ks = r % ksteps;
  const int lane = threadIdx.x & 31;
  const bool hi = lane >= 16;
  const int m = min(mt * 16 + (lane & 15), OC - 1);
#pragma unroll
  for (int j = 0; j < 16; ++j) {
    int ka = ks * 32 + j + (hi ? 16 : 0);
    int ti = ka >> 5, ic = ka & 31;
    int tau = par ? (ti ? 0 : 2) : (ti ? 1 : 3);     // wt[oc,ic,t]=w[ic,oc,3-t]
    float v = (ka < K) ? w[(ic * OC + m) * 4 + tau] : 0.0f;
    af[((size_t)f * 32 + lane) * 16 + j] = (_Float16)v;
  }
}

__global__ void pack_a_mu_kernel(const float* __restrict__ muw,  // [512][512]
                                 _Float16* __restrict__ af) {
  const int f = blockIdx.x * (blockDim.x >> 5) + (threadIdx.x >> 5);
  if (f >= 32 * 16) return;
  const int mt = f / 16, ks = f % 16;
  const int lane = threadIdx.x & 31;
  const bool hi = lane >= 16;
  const int l = mt * 16 + (lane & 15);
#pragma unroll
  for (int j = 0; j < 16; ++j) {
    int ka = ks * 32 + j + (hi ? 16 : 0);
    af[((size_t)f * 32 + lane) * 16 + j] = (_Float16)muw[(size_t)l * LDIM + ka];
  }
}

// ===========================================================================
// helpers
// ===========================================================================
__global__ void pack_x_kernel(const float* __restrict__ x, _Float16* __restrict__ xT,
                              int Bc, int b_origin) {
  int idx = blockIdx.x * blockDim.x + threadIdx.x;
  if (idx >= LDIM * Bc) return;
  int bc = idx / LDIM, p = idx % LDIM;              // coalesced read
  xT[p * Bc + bc] = (_Float16)x[(size_t)(b_origin + bc) * LDIM + p];
}

// zero halo positions 0 and Ppad-1 of a [Ppad][Bc][32] buffer
__global__ void zero_halo_kernel(_Float16* __restrict__ buf, int Ppad, int Bc) {
  int idx = blockIdx.x * blockDim.x + threadIdx.x;
  int n = Bc * 32;
  if (idx >= 2 * n) return;
  size_t pos = (idx < n) ? 0 : (size_t)(Ppad - 1);
  buf[pos * n + (idx % n)] = (_Float16)0.0f;
}

// enc layer0: 1->32ch, k=4, s=2 (0.27 GMAC VALU). One thread per (p,b),
// all 32 out-channels -> contiguous 64B store into [258][Bc][32].
__global__ void conv0_kernel(const _Float16* __restrict__ xT,   // [512][Bc]
                             const float* __restrict__ w,       // [32][4]
                             const float* __restrict__ bias,
                             _Float16* __restrict__ h0,         // [258][Bc][32]
                             int Bc) {
  int idx = blockIdx.x * blockDim.x + threadIdx.x;
  if (idx >= 256 * Bc) return;
  int b = idx % Bc, p = idx / Bc;
  float xv[4];
#pragma unroll
  for (int t = 0; t < 4; ++t) {
    int pin = 2 * p - 1 + t;
    int pc = min(max(pin, 0), 511);
    float v = (float)xT[pc * Bc + b];
    xv[t] = (pin == pc) ? v : 0.0f;
  }
  v16h o0, o1;
#pragma unroll
  for (int oc = 0; oc < 32; ++oc) {
    float acc = bias[oc];
#pragma unroll
    for (int t = 0; t < 4; ++t) acc += w[oc * 4 + t] * xv[t];
    acc = acc > 0.0f ? acc : 0.0f;
    if (oc < 16) o0[oc] = (_Float16)acc; else o1[oc - 16] = (_Float16)acc;
  }
  _Float16* dst = h0 + ((size_t)(p + 1) * Bc + b) * 32;
  *(v16h*)dst = o0;
  *(v16h*)(dst + 16) = o1;
}

// dec layer0: 1->32ch transposed conv (0.07 GMAC VALU), same output scheme.
__global__ void dec0_kernel(const _Float16* __restrict__ din,   // [64][Bc]
                            const float* __restrict__ w,        // [1][32][4]
                            const float* __restrict__ bias,
                            _Float16* __restrict__ g0,          // [130][Bc][32]
                            int Bc) {
  int idx = blockIdx.x * blockDim.x + threadIdx.x;
  if (idx >= 128 * Bc) return;
  int b = idx % Bc, q = idx / Bc;
  int tau0, tau1, pin0, pin1;
  if ((q & 1) == 0) { tau0 = 3; tau1 = 1; pin0 = q / 2 - 1; pin1 = q / 2; }
  else              { tau0 = 2; tau1 = 0; pin0 = (q - 1) / 2; pin1 = (q + 1) / 2; }
  int c0 = min(max(pin0, 0), 63), c1 = min(max(pin1, 0), 63);
  float v0 = (float)din[c0 * Bc + b], v1 = (float)din[c1 * Bc + b];
  v0 = (pin0 == c0) ? v0 : 0.0f;
  v1 = (pin1 == c1) ? v1 : 0.0f;
  v16h o0, o1;
#pragma unroll
  for (int oc = 0; oc < 32; ++oc) {
    float acc = bias[oc] + w[oc * 4 + tau0] * v0 + w[oc * 4 + tau1] * v1;
    acc = acc > 0.0f ? acc : 0.0f;
    if (oc < 16) o0[oc] = (_Float16)acc; else o1[oc - 16] = (_Float16)acc;
  }
  _Float16* dst = g0 + ((size_t)(q + 1) * Bc + b) * 32;
  *(v16h*)dst = o0;
  *(v16h*)(dst + 16) = o1;
}

// ===========================================================================
// WMMA conv layer. Input [Pinp][Bc][32] (zero halo), k = tap*32 + ic, so each
// k-step's B fragment is ONE contiguous v16h per lane.
// ===========================================================================
template <int MT, int KS>
__global__ void conv_wmma_kernel(const _Float16* __restrict__ in,  // [Pinp][Bc][32]
                                 const _Float16* __restrict__ af,
                                 const float* __restrict__ bias,
                                 _Float16* __restrict__ out,       // [Pout+2][Bc][32] or null
                                 float* __restrict__ zout,         // optional f32 [b][Pout]
                                 int Pout, int OC, int Bc, int b_origin) {
  const int wid = blockIdx.x * (blockDim.x >> 5) + (threadIdx.x >> 5);
  const int nbt = Bc >> 4;
  if (wid >= Pout * nbt) return;                     // wave-uniform
  const int p = wid / nbt;
  const int b0 = (wid % nbt) << 4;
  const int lane = threadIdx.x & 31;
  const bool hi = lane >= 16;
  const int ln = lane & 15;
  // per-lane source: channel block (hi?16:0) at (pos, b0+ln)
  const _Float16* inb = in + ((size_t)b0 + ln) * 32 + (hi ? 16 : 0);
  const size_t prow = (size_t)Bc * 32;

  v16h bmv[KS];
#pragma unroll
  for (int ks = 0; ks < KS; ++ks)                     // pos = 2p + ks (padded)
    bmv[ks] = *(const v16h*)(inb + (size_t)(2 * p + ks) * prow);

#pragma unroll
  for (int mt = 0; mt < MT; ++mt) {
    v8f acc = {};
#pragma unroll
    for (int ks = 0; ks < KS; ++ks) {
      v16h a = *(const v16h*)(af + ((size_t)(mt * KS + ks) * 32 + lane) * 16);
      acc = __builtin_amdgcn_wmma_f32_16x16x32_f16(false, a, false, bmv[ks],
                                                   (short)0, acc, false, false);
    }
#pragma unroll
    for (int i = 0; i < 8; ++i) {
      int m = hi ? i + 8 : i;
      int oc = mt * 16 + m;
      int bb = b0 + ln;
      if (oc < OC) {
        float v = acc[i] + bias[oc];
        v = v > 0.0f ? v : 0.0f;
        if (out)  out[((size_t)(p + 1) * Bc + bb) * 32 + oc] = (_Float16)v;
        if (zout && oc == 0) zout[(size_t)(b_origin + bb) * Pout + p] = v;
      }
    }
  }
}

// ===========================================================================
// WMMA transposed-conv layer: 2 live taps -> KS=2, k = tap*32 + ic.
// outT (optional): single-channel transposed output [b][Pout] for mu.
// ===========================================================================
template <int MT, int KS>
__global__ void deconv_wmma_kernel(const _Float16* __restrict__ in,  // [Pinp][Bc][32]
                                   const _Float16* __restrict__ af,  // x2 parity
                                   const float* __restrict__ bias,
                                   _Float16* __restrict__ out,       // [Pout+2][Bc][32] or null
                                   _Float16* __restrict__ outT,      // [b][Pout] or null
                                   int Pout, int OC, int Bc) {
  const int wid = blockIdx.x * (blockDim.x >> 5) + (threadIdx.x >> 5);
  const int nbt = Bc >> 4;
  if (wid >= Pout * nbt) return;
  const int q = wid / nbt;
  const int b0 = (wid % nbt) << 4;
  const int lane = threadIdx.x & 31;
  const bool hi = lane >= 16;
  const int ln = lane & 15;
  const int par = q & 1;
  const int pin0p = (q + par) >> 1;                  // padded tap positions
  const _Float16* inb = in + ((size_t)b0 + ln) * 32 + (hi ? 16 : 0);
  const size_t prow = (size_t)Bc * 32;
  const _Float16* afp = af + (size_t)par * MT * KS * 32 * 16;

  v16h bmv[KS];
#pragma unroll
  for (int ks = 0; ks < KS; ++ks)
    bmv[ks] = *(const v16h*)(inb + (size_t)(pin0p + ks) * prow);

#pragma unroll
  for (int mt = 0; mt < MT; ++mt) {
    v8f acc = {};
#pragma unroll
    for (int ks = 0; ks < KS; ++ks) {
      v16h a = *(const v16h*)(afp + ((size_t)(mt * KS + ks) * 32 + lane) * 16);
      acc = __builtin_amdgcn_wmma_f32_16x16x32_f16(false, a, false, bmv[ks],
                                                   (short)0, acc, false, false);
    }
#pragma unroll
    for (int i = 0; i < 8; ++i) {
      int m = hi ? i + 8 : i;
      int oc = mt * 16 + m;
      int bb = b0 + ln;
      if (oc < OC) {
        float v = acc[i] + bias[oc];
        v = v > 0.0f ? v : 0.0f;
        if (out)  out[((size_t)(q + 1) * Bc + bb) * 32 + oc] = (_Float16)v;
        if (outT && oc == 0) outT[(size_t)bb * Pout + q] = (_Float16)v;
      }
    }
  }
}

// ===========================================================================
// VQ: argmin + code gather + embedding matvec (2.5% of FLOPs, VALU).
// ===========================================================================
__global__ void vq_kernel(const float* __restrict__ zall, const float* __restrict__ cb,
                          const float* __restrict__ embw, const float* __restrict__ embb,
                          const int* __restrict__ spk,
                          float* __restrict__ codeout, _Float16* __restrict__ decin,
                          int Bc, int b_origin) {
  __shared__ float zsh[8][DDIM];
  const int wave = threadIdx.x >> 5;
  const int lane = threadIdx.x & 31;
  const int bc = blockIdx.x * 8 + wave;
  const int b = b_origin + bc;
  const float* z = zall + (size_t)b * DDIM;
  zsh[wave][lane] = z[lane];
  zsh[wave][lane + 32] = z[lane + 32];
  __syncthreads();

  float bestv = 3.0e38f;
  int bestk = 0;
  for (int k = lane; k < KCB; k += 32) {
    const float* c = cb + (size_t)k * DDIM;
    float s = 0.0f, cc = 0.0f;
#pragma unroll 8
    for (int d = 0; d < DDIM; ++d) { s += zsh[wave][d] * c[d]; cc += c[d] * c[d]; }
    float v = 0.5f * cc - s;                        // 0.5||z||^2 const per row
    if (v < bestv) { bestv = v; bestk = k; }        // ascending k -> first min
  }
#pragma unroll
  for (int off = 16; off >= 1; off >>= 1) {
    float ov = __shfl_down(bestv, off);
    int ok = __shfl_down(bestk, off);
    if (ov < bestv || (ov == bestv && ok < bestk)) { bestv = ov; bestk = ok; }
  }
  int best = __shfl(bestk, 0);

  const float* c = cb + (size_t)best * DDIM;
  for (int d = lane; d < DDIM; d += 32)
    codeout[(size_t)b * DDIM + d] = c[d];
  int sp = spk[b];
  for (int j = lane; j < DDIM; j += 32) {
    const float* er = embw + (size_t)j * (YDIM + DDIM);
    float acc = embb[j] + er[DDIM + sp];
#pragma unroll 8
    for (int d = 0; d < DDIM; ++d) acc += c[d] * er[d];
    decin[j * Bc + bc] = (_Float16)acc;
  }
}

// ===========================================================================
// mu GEMM: out[b][l] = sum_p g2T[b][p]*mu_w[l][p] + mu_b[l]; B frag = one v16h.
// ===========================================================================
__global__ void mu_kernel(const _Float16* __restrict__ g2T,  // [b][512]
                          const _Float16* __restrict__ af,
                          const float* __restrict__ mub,
                          float* __restrict__ outr, int Bc, int b_origin) {
  const int wid = blockIdx.x * (blockDim.x >> 5) + (threadIdx.x >> 5);
  const int nbt = Bc >> 4;
  if (wid >= 32 * nbt) return;
  const int mt = wid / nbt;
  const int b0 = (wid % nbt) << 4;
  const int lane = threadIdx.x & 31;
  const bool hi = lane >= 16;
  const int ln = lane & 15;
  const _Float16* g2r = g2T + (size_t)(b0 + ln) * LDIM + (hi ? 16 : 0);

  v8f acc = {};
#pragma unroll
  for (int ks = 0; ks < 16; ++ks) {
    v16h a = *(const v16h*)(af + ((size_t)(mt * 16 + ks) * 32 + lane) * 16);
    v16h bm = *(const v16h*)(g2r + ks * 32);
    acc = __builtin_amdgcn_wmma_f32_16x16x32_f16(false, a, false, bm,
                                                 (short)0, acc, false, false);
  }
#pragma unroll
  for (int i = 0; i < 8; ++i) {
    int m = hi ? i + 8 : i;
    int l = mt * 16 + m;
    outr[(size_t)(b_origin + b0 + ln) * LDIM + l] = acc[i] + mub[l];
  }
}

// ===========================================================================
static inline size_t alignup(size_t v) { return (v + 255) & ~(size_t)255; }

extern "C" void kernel_launch(void* const* d_in, const int* in_sizes, int n_in,
                              void* d_out, int out_size, void* d_ws, size_t ws_size,
                              hipStream_t stream) {
  const float* x    = (const float*)d_in[0];
  const float* ew0  = (const float*)d_in[1];
  const float* eb0  = (const float*)d_in[2];
  const float* ew1  = (const float*)d_in[3];
  const float* eb1  = (const float*)d_in[4];
  const float* ew2  = (const float*)d_in[5];
  const float* eb2  = (const float*)d_in[6];
  const float* cb   = (const float*)d_in[7];
  const float* embw = (const float*)d_in[8];
  const float* embb = (const float*)d_in[9];
  const float* dw0  = (const float*)d_in[10];
  const float* db0  = (const float*)d_in[11];
  const float* dw1  = (const float*)d_in[12];
  const float* db1  = (const float*)d_in[13];
  const float* dw2  = (const float*)d_in[14];
  const float* db2  = (const float*)d_in[15];
  const float* muw  = (const float*)d_in[16];
  const float* mub  = (const float*)d_in[17];
  const int*   spk  = (const int*)d_in[18];

  float* zreg    = (float*)d_out;
  float* codereg = zreg + (size_t)BTOT * DDIM;
  float* outreg  = codereg + (size_t)BTOT * DDIM;

  const size_t F = 32 * 16;                 // halfs per fragment
  const size_t nf_c1 = 2 * 4, nf_c2 = 1 * 4, nf_d1 = 2 * 2 * 2, nf_d2 = 2 * 1 * 2,
               nf_mu = 32 * 16;
  size_t abytes = alignup((nf_c1 + nf_c2 + nf_d1 + nf_d2 + nf_mu) * F * 2);

  int Bc = BTOT;
  for (;;) {
    size_t need = abytes
                + alignup((size_t)LDIM * Bc * 2)          // xT
                + alignup((size_t)258 * Bc * 32 * 2)      // bufA: h0 / g1
                + alignup((size_t)130 * Bc * 32 * 2)      // bufB: h1 / g0
                + alignup((size_t)DDIM * Bc * 2)          // decin
                + alignup((size_t)Bc * LDIM * 2);         // g2T
    if (need <= ws_size || Bc <= 16) break;
    Bc >>= 1;
  }

  char* wsp = (char*)d_ws;
  _Float16* afc1 = (_Float16*)wsp;
  _Float16* afc2 = afc1 + nf_c1 * F;
  _Float16* afd1 = afc2 + nf_c2 * F;
  _Float16* afd2 = afd1 + nf_d1 * F;
  _Float16* afmu = afd2 + nf_d2 * F;
  wsp += abytes;
  _Float16* xT    = (_Float16*)wsp; wsp += alignup((size_t)LDIM * Bc * 2);
  _Float16* bufA  = (_Float16*)wsp; wsp += alignup((size_t)258 * Bc * 32 * 2);
  _Float16* bufB  = (_Float16*)wsp; wsp += alignup((size_t)130 * Bc * 32 * 2);
  _Float16* decin = (_Float16*)wsp; wsp += alignup((size_t)DDIM * Bc * 2);
  _Float16* g2T   = (_Float16*)wsp;

  pack_a_enc_kernel<<<1, 256, 0, stream>>>(ew1, afc1, 32, 32, 128, 4, 2);
  pack_a_enc_kernel<<<1, 256, 0, stream>>>(ew2, afc2, 1, 32, 128, 4, 1);
  pack_a_dec_kernel<<<1, 256, 0, stream>>>(dw1, afd1, 32, 64, 2, 2);
  pack_a_dec_kernel<<<1, 256, 0, stream>>>(dw2, afd2, 1, 64, 2, 1);
  pack_a_mu_kernel<<<64, 256, 0, stream>>>(muw, afmu);

  const int nbt = Bc >> 4;
  auto wblocks = [](int waves) { return (unsigned)((waves + 7) / 8); };
  auto tblocks = [](int thr) { return (unsigned)((thr + 255) / 256); };

  for (int b0 = 0; b0 < BTOT; b0 += Bc) {
    pack_x_kernel<<<tblocks(LDIM * Bc), 256, 0, stream>>>(x, xT, Bc, b0);
    // ---- encoder ----
    zero_halo_kernel<<<tblocks(2 * Bc * 32), 256, 0, stream>>>(bufA, 258, Bc);
    conv0_kernel<<<tblocks(256 * Bc), 256, 0, stream>>>(xT, ew0, eb0, bufA, Bc);
    zero_halo_kernel<<<tblocks(2 * Bc * 32), 256, 0, stream>>>(bufB, 130, Bc);
    conv_wmma_kernel<2, 4><<<wblocks(128 * nbt), 256, 0, stream>>>(
        bufA, afc1, eb1, bufB, nullptr, 128, 32, Bc, b0);
    conv_wmma_kernel<1, 4><<<wblocks(64 * nbt), 256, 0, stream>>>(
        bufB, afc2, eb2, nullptr, zreg, 64, 1, Bc, b0);
    // ---- VQ + embedding ----
    vq_kernel<<<(unsigned)(Bc / 8), 256, 0, stream>>>(
        zreg, cb, embw, embb, spk, codereg, decin, Bc, b0);
    // ---- decoder (bufB = g0, bufA = g1) ----
    zero_halo_kernel<<<tblocks(2 * Bc * 32), 256, 0, stream>>>(bufB, 130, Bc);
    dec0_kernel<<<tblocks(128 * Bc), 256, 0, stream>>>(decin, dw0, db0, bufB, Bc);
    zero_halo_kernel<<<tblocks(2 * Bc * 32), 256, 0, stream>>>(bufA, 258, Bc);
    deconv_wmma_kernel<2, 2><<<wblocks(256 * nbt), 256, 0, stream>>>(
        bufB, afd1, db1, bufA, nullptr, 256, 32, Bc);
    deconv_wmma_kernel<1, 2><<<wblocks(512 * nbt), 256, 0, stream>>>(
        bufA, afd2, db2, nullptr, g2T, 512, 1, Bc);
    // ---- final linear ----
    mu_kernel<<<wblocks(32 * nbt), 256, 0, stream>>>(g2T, afmu, mub, outreg, Bc, b0);
  }
  (void)in_sizes; (void)n_in; (void)out_size;
}
